// OnlineLSTM_89850715832463
// MI455X (gfx1250) — matrix operand order repaired and checked by reading, not compiled, not run on hardware
//
#include <hip/hip_runtime.h>

// ---------------------------------------------------------------------------
// OnlineLSTM for MI455X (gfx1250, wave32, WMMA, async-to-LDS).
//   K1: fp32 -> bf16 conversion (inp, W_ih, W_hh) + global_prefetch
//   K2: xg = inp @ W_ih^T + b_ih   (v_wmma_f32_16x16x32_bf16, 32x32 per wave)
//   K3: persistent LSTM scan: W_hh slice pinned in 128KB LDS per WGP via
//       global_load_async_to_lds_b128; per-step h broadcast also async;
//       per-step matvec via WMMA with h broadcast into all B columns;
//       grid-wide generation barrier between steps.
//   K4: out = tanh(h_fin @ fc_w^T + fc_b)
// ---------------------------------------------------------------------------

typedef __attribute__((ext_vector_type(16))) __bf16 bf16x16;
typedef __attribute__((ext_vector_type(8)))  __bf16 bf16x8;
typedef __attribute__((ext_vector_type(8)))  float  f32x8;

namespace {
constexpr int T_WIN    = 2048;
constexpr int K_IN     = 4096;
constexpr int HID      = 1024;
constexpr int G4       = 4096;          // 4*HID gate rows
constexpr int METRIC_N = 32;
constexpr int STRIDE_N = 1024;

constexpr int SCAN_BLOCKS  = 64;        // persistent blocks; 16 hidden units each
constexpr int SCAN_THREADS = 128;       // 4 waves -> one 16-row gate tile per wave

// workspace byte offsets (256B aligned)
constexpr size_t OFF_INP_BF = 0;                                          // 16 MB
constexpr size_t OFF_WIH_BF = OFF_INP_BF + (size_t)T_WIN * K_IN * 2;      // 32 MB
constexpr size_t OFF_WHH_BF = OFF_WIH_BF + (size_t)G4 * K_IN * 2;        // 8 MB
constexpr size_t OFF_XG     = OFF_WHH_BF + (size_t)G4 * HID * 2;         // 32 MB
constexpr size_t OFF_H32    = OFF_XG + (size_t)T_WIN * G4 * 4;
constexpr size_t OFF_C32    = OFF_H32 + 4096;
constexpr size_t OFF_HBF    = OFF_C32 + 4096;
constexpr size_t OFF_BAR    = OFF_HBF + 4096;

constexpr int SCAN_SHMEM = 64 * HID * 2   /* W_hh slice (64 rows bf16) */
                         + HID * 2        /* h broadcast (bf16)        */
                         + 64 * 4         /* gate pre-activations      */
                         + 16 * 4;        /* c state                   */
} // namespace

// ---- CDNA5 async memory->LDS helpers (CDNA5 ISA) -----------

// LDS offset = low 32 bits of a generic pointer (aperture is in bits 63:32).
__device__ __forceinline__ unsigned lds_off(const void* p) {
  return (unsigned)(unsigned long long)p;
}

// One 16-byte async copy memory -> LDS per active lane (ASYNCcnt-tracked).
__device__ __forceinline__ void async_copy_b128(void* lds_dst, const void* gsrc) {
  asm volatile("global_load_async_to_lds_b128 %0, %1, off"
               :: "v"(lds_off(lds_dst)), "v"(gsrc)
               : "memory");
}

__device__ __forceinline__ void wait_async0() {
  asm volatile("s_wait_asynccnt 0x0" ::: "memory");
}

// ---- WMMA fragment helpers (layouts per CDNA5 ISA 7.12.2) ------------------

__device__ __forceinline__ bf16x16 cat8(bf16x8 lo, bf16x8 hi) {
  return __builtin_shufflevector(lo, hi, 0,1,2,3,4,5,6,7,8,9,10,11,12,13,14,15);
}

// A-matrix 16x32 bf16: lane = M row; elems = 16 contiguous K (kb + 16*(lane>>4))
__device__ __forceinline__ bf16x16 frag_a(const __bf16* __restrict__ base, int ld,
                                          int row0, int kc) {
  const int lane = threadIdx.x & 31;
  const __bf16* p = base + (size_t)(row0 + (lane & 15)) * ld + kc + ((lane >> 4) << 4);
  return cat8(*(const bf16x8*)p, *(const bf16x8*)(p + 8));
}

// B-matrix 32x16 bf16 from row-major W[N][K]: lane = N col; elems = 16 contiguous K
__device__ __forceinline__ bf16x16 frag_b(const __bf16* __restrict__ base, int ld,
                                          int col0, int kc) {
  const int lane = threadIdx.x & 31;
  const __bf16* p = base + (size_t)(col0 + (lane & 15)) * ld + kc + ((lane >> 4) << 4);
  return cat8(*(const bf16x8*)p, *(const bf16x8*)(p + 8));
}

// B-matrix with vector v broadcast into every column (matvec via WMMA)
__device__ __forceinline__ bf16x16 frag_b_bcast(const __bf16* __restrict__ v, int kc) {
  const int lane = threadIdx.x & 31;
  const __bf16* p = v + kc + ((lane >> 4) << 4);
  return cat8(*(const bf16x8*)p, *(const bf16x8*)(p + 8));
}

__device__ __forceinline__ f32x8 wmma_bf16(bf16x16 a, bf16x16 b, f32x8 c) {
  return __builtin_amdgcn_wmma_f32_16x16x32_bf16(false, a, false, b, (short)0, c,
                                                 false, false);
}

__device__ __forceinline__ float sigmoidf(float x) {
  return 1.0f / (1.0f + __expf(-x));
}

// ---- K1: fp32 -> bf16 (streaming, with prefetch) ---------------------------

__global__ void cvt_f32_bf16(const float* __restrict__ in, __bf16* __restrict__ out,
                             int n) {
  int i = blockIdx.x * blockDim.x + threadIdx.x;
  const int stride = gridDim.x * blockDim.x;
  for (; i < n; i += stride) {
    if (i + stride < n) __builtin_prefetch(in + i + stride, 0, 0);
    out[i] = (__bf16)in[i];
  }
}

// ---- K2: xg = inp @ W_ih^T + b_ih -----------------------------------------
// block = 8 waves (2x4); each wave computes a 32x32 C patch (2x2 WMMA tiles).
// Block tile: 64(M) x 128(N). Grid: (4096/128, 2048/64) = (32, 32).

__global__ __launch_bounds__(256)
void gemm_xg(const __bf16* __restrict__ A,   // [T_WIN][K_IN] bf16
             const __bf16* __restrict__ W,   // [G4][K_IN]    bf16 (row-major N,K)
             const float*  __restrict__ bih, // [G4]
             float* __restrict__ xg) {       // [T_WIN][G4]
  const int lane = threadIdx.x & 31;
  const int wave = threadIdx.x >> 5;
  const int m0 = blockIdx.y * 64  + (wave & 1) * 32;
  const int n0 = blockIdx.x * 128 + (wave >> 1) * 32;

  f32x8 acc00 = {}, acc01 = {}, acc10 = {}, acc11 = {};
  for (int kc = 0; kc < K_IN; kc += 32) {
    bf16x16 a0 = frag_a(A, K_IN, m0,      kc);
    bf16x16 a1 = frag_a(A, K_IN, m0 + 16, kc);
    bf16x16 b0 = frag_b(W, K_IN, n0,      kc);
    bf16x16 b1 = frag_b(W, K_IN, n0 + 16, kc);
    acc00 = wmma_bf16(a0, b0, acc00);
    acc01 = wmma_bf16(a0, b1, acc01);
    acc10 = wmma_bf16(a1, b0, acc10);
    acc11 = wmma_bf16(a1, b1, acc11);
  }

  const int row = (lane >> 4) << 3;   // 0 or 8
  const int col = lane & 15;
#pragma unroll
  for (int r = 0; r < 8; ++r) {
    int m = m0 + row + r;
    xg[(size_t)m * G4 + (n0 + col)]        = acc00[r] + bih[n0 + col];
    xg[(size_t)m * G4 + (n0 + 16 + col)]   = acc01[r] + bih[n0 + 16 + col];
    m += 16;
    xg[(size_t)m * G4 + (n0 + col)]        = acc10[r] + bih[n0 + col];
    xg[(size_t)m * G4 + (n0 + 16 + col)]   = acc11[r] + bih[n0 + 16 + col];
  }
}

// ---- K3 init: state + barrier reset (every launch, graph-safe) -------------

__global__ void scan_init(const float* __restrict__ h1, const float* __restrict__ c1,
                          float* __restrict__ h32, float* __restrict__ c32,
                          __bf16* __restrict__ hbf, int* __restrict__ bar) {
  const int i = threadIdx.x;   // 1024 threads
  if (i < 2) bar[i] = 0;
  h32[i] = h1[i];
  c32[i] = c1[i];
  hbf[i] = (__bf16)h1[i];
}

// ---- K3: persistent LSTM scan ---------------------------------------------
// Block b owns hidden units j in [16b, 16b+16) -> 64 W_hh gate rows, pinned
// in LDS (128KB of the WGP's 320KB) via async-to-LDS. Wave w computes gate
// w's 16-row tile each step; h is broadcast into LDS asynchronously.

__global__ __launch_bounds__(SCAN_THREADS)
void lstm_scan(const __bf16* __restrict__ Whh,  // [G4][HID] bf16
               const float*  __restrict__ xg,   // [T_WIN][G4]
               const float*  __restrict__ bhh,  // [G4]
               float*  __restrict__ h32,        // [HID] live h (fp32)
               float*  __restrict__ c32,        // [HID] initial c (fp32)
               __bf16* __restrict__ hbf,        // [HID] live h (bf16)
               int*    __restrict__ bar,        // [0]=count [1]=generation
               float*  __restrict__ out) {      // d_out: [32 | h_last | c_last]
  extern __shared__ char smem[];
  __bf16* sW = (__bf16*)smem;                              // [64][HID]
  __bf16* sH = (__bf16*)(smem + 64 * HID * 2);             // [HID]
  float*  sG = (float*)(smem + 64 * HID * 2 + HID * 2);    // [64]
  float*  sC = sG + 64;                                    // [16]

  const int tid  = threadIdx.x;
  const int lane = tid & 31;
  const int wave = tid >> 5;          // 0..3 == gate index
  const int j0   = blockIdx.x * 16;   // first hidden unit owned by this block

  // Pin this block's 64 W_hh rows into LDS without touching VGPRs:
  // 8192 x 16B async copies (64 per thread), ASYNCcnt-tracked.
  for (int idx = tid; idx < 64 * (HID / 8); idx += SCAN_THREADS) {
    const int lr = idx >> 7;            // local row  (gate*16 + u)
    const int kk = (idx & 127) << 3;    // K offset (elements)
    const int g = lr >> 4, u = lr & 15;
    async_copy_b128(sW + (size_t)lr * HID + kk,
                    Whh + (size_t)(g * HID + j0 + u) * HID + kk);
  }
  if (tid < 16) sC[tid] = c32[j0 + tid];   // overlaps with the async DMA
  wait_async0();
  __syncthreads();

  for (int t = 0; t < T_WIN; ++t) {
    // Broadcast current h (2KB bf16) into LDS: one async b128 per thread.
    async_copy_b128(sH + tid * 8, hbf + tid * 8);
    wait_async0();
    __syncthreads();

    // Gate tile: acc = W_slice(16x1024) * h, h replicated in all 16 B columns.
    f32x8 acc = {};
    for (int kc = 0; kc < HID; kc += 32) {
      bf16x16 a = frag_a(sW, HID, wave * 16, kc);
      bf16x16 b = frag_b_bcast(sH, kc);
      acc = wmma_bf16(a, b, acc);
    }
    if ((lane & 15) == 0) {           // column 0 of C: rows r + 8*(lane>=16)
      const int half = (lane >> 4) << 3;
#pragma unroll
      for (int r = 0; r < 8; ++r) sG[wave * 16 + half + r] = acc[r];
    }
    __syncthreads();

    // Elementwise LSTM cell update for the 16 owned units.
    if (tid < 16) {
      const int j = j0 + tid;
      const size_t xrow = (size_t)t * G4;
      const float ig = sG[0  + tid] + xg[xrow + 0 * HID + j] + bhh[0 * HID + j];
      const float fg = sG[16 + tid] + xg[xrow + 1 * HID + j] + bhh[1 * HID + j];
      const float gg = sG[32 + tid] + xg[xrow + 2 * HID + j] + bhh[2 * HID + j];
      const float og = sG[48 + tid] + xg[xrow + 3 * HID + j] + bhh[3 * HID + j];
      const float cn = sigmoidf(fg) * sC[tid] + sigmoidf(ig) * tanhf(gg);
      const float hn = sigmoidf(og) * tanhf(cn);
      sC[tid] = cn;
      h32[j] = hn;
      hbf[j] = (__bf16)hn;
      if (t == STRIDE_N - 1) {        // detached h_last / c_last
        out[METRIC_N + j]       = hn;
        out[METRIC_N + HID + j] = cn;
      }
    }

    // Grid-wide generation barrier (64 persistent blocks).
    __syncthreads();
    if (tid == 0) {
      __threadfence();
      if (atomicAdd(bar, 1) == SCAN_BLOCKS - 1) {
        bar[0] = 0;
        __threadfence();
        ((volatile int*)bar)[1] = t + 1;
      } else {
        while (((volatile int*)bar)[1] < t + 1) __builtin_amdgcn_s_sleep(2);
      }
    }
    __syncthreads();
  }
}

// ---- K4: out = tanh(h_fin @ fc_w^T + fc_b) --------------------------------

__global__ void fc_out(const float* __restrict__ h, const float* __restrict__ fcw,
                       const float* __restrict__ fcb, float* __restrict__ out) {
  const int o = threadIdx.x;
  if (o < METRIC_N) {
    float s = fcb[o];
    for (int k = 0; k < HID; ++k) s += h[k] * fcw[(size_t)o * HID + k];
    out[o] = tanhf(s);
  }
}

// ---------------------------------------------------------------------------

extern "C" void kernel_launch(void* const* d_in, const int* in_sizes, int n_in,
                              void* d_out, int out_size, void* d_ws, size_t ws_size,
                              hipStream_t stream) {
  (void)in_sizes; (void)n_in; (void)out_size; (void)ws_size;
  const float* inp  = (const float*)d_in[0];
  const float* h1   = (const float*)d_in[1];
  const float* c1   = (const float*)d_in[2];
  const float* W_ih = (const float*)d_in[3];
  const float* W_hh = (const float*)d_in[4];
  const float* b_ih = (const float*)d_in[5];
  const float* b_hh = (const float*)d_in[6];
  const float* fc_w = (const float*)d_in[7];
  const float* fc_b = (const float*)d_in[8];
  float* out = (float*)d_out;

  char* ws = (char*)d_ws;
  __bf16* inp_bf = (__bf16*)(ws + OFF_INP_BF);
  __bf16* wih_bf = (__bf16*)(ws + OFF_WIH_BF);
  __bf16* whh_bf = (__bf16*)(ws + OFF_WHH_BF);
  float*  xg     = (float*)(ws + OFF_XG);
  float*  h32    = (float*)(ws + OFF_H32);
  float*  c32    = (float*)(ws + OFF_C32);
  __bf16* hbf    = (__bf16*)(ws + OFF_HBF);
  int*    bar    = (int*)(ws + OFF_BAR);

  // allow >64KB dynamic LDS for the scan kernel (host-side attr, capture-safe)
  (void)hipFuncSetAttribute((const void*)lstm_scan,
                            hipFuncAttributeMaxDynamicSharedMemorySize, SCAN_SHMEM);

  // K1: conversions
  cvt_f32_bf16<<<4096, 256, 0, stream>>>(inp,  inp_bf, T_WIN * K_IN);
  cvt_f32_bf16<<<4096, 256, 0, stream>>>(W_ih, wih_bf, G4 * K_IN);
  cvt_f32_bf16<<<4096, 256, 0, stream>>>(W_hh, whh_bf, G4 * HID);

  // K2: input-side gate GEMM
  gemm_xg<<<dim3(G4 / 128, T_WIN / 64), 256, 0, stream>>>(inp_bf, wih_bf, b_ih, xg);

  // K3: state init + persistent scan
  scan_init<<<1, 1024, 0, stream>>>(h1, c1, h32, c32, hbf, bar);
  lstm_scan<<<SCAN_BLOCKS, SCAN_THREADS, SCAN_SHMEM, stream>>>(
      whh_bf, xg, b_hh, h32, c32, hbf, bar, out);

  // K4: final projection
  fc_out<<<1, 32, 0, stream>>>(h32, fc_w, fc_b, out);
}
